// GraphPatchMixer_52072183497307
// MI455X (gfx1250) — matrix operand order repaired and checked by reading, not compiled, not run on hardware
//
#include <hip/hip_runtime.h>
#include <math.h>

// ---------------- types for WMMA / async copy ----------------
typedef __attribute__((ext_vector_type(16))) __bf16 v16bf;
typedef __attribute__((ext_vector_type(8)))  __bf16 v8bf;
typedef __attribute__((ext_vector_type(8)))  float  v8f;
typedef __attribute__((ext_vector_type(4)))  int    v4i;

#define GLOBAL_AS __attribute__((address_space(1)))
#define LDS_AS    __attribute__((address_space(3)))

// Problem constants (match reference)
#define BB 2
#define NN 8192
#define DD 256
#define HEADS 4
#define HEAD_DIM 64
#define KNN 9
#define RTOT (BB * NN)          // 16384 rows
#define QKVW (3 * DD)           // 768
#define WSTRIDE 272             // padded LDS row stride (bf16 elems): 544B, spreads banks

// ---------------- f32 -> bf16 conversion (8 elems / thread) ----------------
__global__ void __launch_bounds__(256) cvt_bf16_kernel(const float* __restrict__ in,
                                                       __bf16* __restrict__ out, int n8) {
    int i = blockIdx.x * blockDim.x + threadIdx.x;
    if (i >= n8) return;
    const float4* p = (const float4*)in + 2 * (size_t)i;
    float4 a = p[0], b = p[1];
    v8bf o;
    o[0] = (__bf16)a.x; o[1] = (__bf16)a.y; o[2] = (__bf16)a.z; o[3] = (__bf16)a.w;
    o[4] = (__bf16)b.x; o[5] = (__bf16)b.y; o[6] = (__bf16)b.z; o[7] = (__bf16)b.w;
    *(v8bf*)(out + 8 * (size_t)i) = o;
}

// ---------------- kNN: per-thread query, coords tiled through LDS ----------------
__global__ void __launch_bounds__(256) knn_kernel(const float* __restrict__ coords,
                                                  const unsigned char* __restrict__ mask,
                                                  int* __restrict__ nn_idx, int N) {
    __shared__ float scx[256], scy[256];
    __shared__ unsigned char smk[256];
    const int b = blockIdx.y;
    const int n = blockIdx.x * 256 + threadIdx.x;
    const float* cb = coords + (size_t)b * N * 2;
    const float qx = cb[2 * n], qy = cb[2 * n + 1];
    const float sqn = qx * qx + qy * qy;

    float bd[KNN];
    int   bi[KNN];
#pragma unroll
    for (int j = 0; j < KNN; ++j) { bd[j] = INFINITY; bi[j] = 0; }

    for (int t = 0; t < N; t += 256) {
        int col = t + threadIdx.x;
        scx[threadIdx.x] = cb[2 * col];
        scy[threadIdx.x] = cb[2 * col + 1];
        smk[threadIdx.x] = mask[(size_t)b * N + col];
        __syncthreads();
        for (int j = 0; j < 256; ++j) {
            float mx = scx[j], my = scy[j];
            float d = sqn + (mx * mx + my * my) - 2.0f * (qx * mx + qy * my);
            if (smk[j]) d = INFINITY;
            if (d < bd[KNN - 1]) {
                bd[KNN - 1] = d; bi[KNN - 1] = t + j;
#pragma unroll
                for (int p = KNN - 1; p > 0; --p) {
                    if (bd[p] < bd[p - 1]) {
                        float td = bd[p]; bd[p] = bd[p - 1]; bd[p - 1] = td;
                        int   ti = bi[p]; bi[p] = bi[p - 1]; bi[p - 1] = ti;
                    }
                }
            }
        }
        __syncthreads();
    }
#pragma unroll
    for (int j = 0; j < KNN; ++j) nn_idx[((size_t)b * N + n) * KNN + j] = bi[j];
}

// ---------------- shared GEMM body: per-block stage W tile into LDS (async),
// per-wave 32x64 tile: 2 A frags x 4 B frags = 8 WMMA per k-step ----------------
__device__ __forceinline__ void stage_w_async(const __bf16* __restrict__ W, int n0,
                                              __bf16* sW, int tid) {
    // copy 64 rows x 256 bf16 (32 x 16B chunks per row) with async global->LDS
#pragma unroll
    for (int i = 0; i < 16; ++i) {
        int c   = tid + i * 128;     // 0..2047
        int row = c >> 5;
        int ch  = c & 31;
        __builtin_amdgcn_global_load_async_to_lds_b128(
            (GLOBAL_AS v4i*)(W + (size_t)(n0 + row) * DD + ch * 8),
            (LDS_AS v4i*)(sW + row * WSTRIDE + ch * 8), 0, 0);
    }
    __builtin_amdgcn_s_wait_asynccnt(0);
    __syncthreads();
}

__device__ __forceinline__ v16bf lds_b_frag(const __bf16* sW, int col, int kb) {
    const __bf16* p = sW + col * WSTRIDE + kb;     // 16B-aligned
    v8bf lo = *(const v8bf*)(p);
    v8bf hi = *(const v8bf*)(p + 8);
    v16bf b;
#pragma unroll
    for (int i = 0; i < 8; ++i) { b[i] = lo[i]; b[8 + i] = hi[i]; }
    return b;
}

__device__ __forceinline__ v16bf a_frag(const __bf16* xrow, int k0, int half) {
    v8bf lo = *(const v8bf*)(xrow + k0 + half * 16);
    v8bf hi = *(const v8bf*)(xrow + k0 + half * 16 + 8);
    v16bf a;
#pragma unroll
    for (int i = 0; i < 8; ++i) { a[i] = lo[i]; a[8 + i] = hi[i]; }
    return a;
}

// GEMM 1: qkv(16384x768) = Xbf(16384x256) * Wqkv^T (cols of W^T = rows of Wqkv)
__global__ void __launch_bounds__(128) gemm_qkv_kernel(const __bf16* __restrict__ X,
                                                       const __bf16* __restrict__ W,
                                                       float* __restrict__ Y) {
    __shared__ __bf16 sW[64 * WSTRIDE];
    const int tid  = threadIdx.x;
    const int lane = tid & 31, wave = tid >> 5;
    const int half = lane >> 4, l16 = lane & 15;
    const int m0 = blockIdx.x * 128 + wave * 32;
    const int n0 = blockIdx.y * 64;

    stage_w_async(W, n0, sW, tid);

    v8f acc[2][4] = {};
    const __bf16* xr0 = X + (size_t)(m0 + l16) * DD;
    const __bf16* xr1 = X + (size_t)(m0 + 16 + l16) * DD;

    for (int k0 = 0; k0 < DD; k0 += 32) {
        if (k0 + 32 < DD) {
            __builtin_prefetch(xr0 + k0 + 32 + half * 8, 0, 0);
            __builtin_prefetch(xr1 + k0 + 32 + half * 8, 0, 0);
        }
        v16bf a0 = a_frag(xr0, k0, half);
        v16bf a1 = a_frag(xr1, k0, half);
        const int kb = k0 + half * 16;
#pragma unroll
        for (int t = 0; t < 4; ++t) {
            v16bf b = lds_b_frag(sW, t * 16 + l16, kb);
            acc[0][t] = __builtin_amdgcn_wmma_f32_16x16x32_bf16(false, a0, false, b, (short)0, acc[0][t], false, false);
            acc[1][t] = __builtin_amdgcn_wmma_f32_16x16x32_bf16(false, a1, false, b, (short)0, acc[1][t], false, false);
        }
    }
#pragma unroll
    for (int s = 0; s < 2; ++s)
#pragma unroll
        for (int v = 0; v < 8; ++v) {
            const size_t r = (size_t)(m0 + s * 16 + half * 8 + v) * QKVW + n0 + l16;
#pragma unroll
            for (int t = 0; t < 4; ++t) Y[r + t * 16] = acc[s][t][v];
        }
}

// ---------------- neighbor attention + GELU, one wave per (node, head) ----------------
__global__ void __launch_bounds__(128) attn_kernel(const float* __restrict__ qkv,
                                                   const int* __restrict__ nn_idx,
                                                   const unsigned char* __restrict__ mask,
                                                   __bf16* __restrict__ gout, int N) {
    const int node = blockIdx.x;          // 0 .. B*N-1
    const int b = node / N;
    const int h = threadIdx.x >> 5;
    const int lane = threadIdx.x & 31;

    const float* qp = qkv + (size_t)node * QKVW + h * HEAD_DIM;
    const float q0 = qp[2 * lane], q1 = qp[2 * lane + 1];

    int   nb[KNN];
    float logit[KNN];
#pragma unroll
    for (int j = 0; j < KNN; ++j) {
        const int idx = nn_idx[(size_t)node * KNN + j];
        nb[j] = idx;
        const float* kp = qkv + (size_t)(b * N + idx) * QKVW + DD + h * HEAD_DIM;
        float p = q0 * kp[2 * lane] + q1 * kp[2 * lane + 1];
#pragma unroll
        for (int o = 16; o > 0; o >>= 1) p += __shfl_xor(p, o, 32);
        logit[j] = (mask[(size_t)b * N + idx] != 0) ? -INFINITY : p * 0.125f; // 1/sqrt(64)
    }
    float mx = -INFINITY;
#pragma unroll
    for (int j = 0; j < KNN; ++j) mx = fmaxf(mx, logit[j]);
    float w[KNN], s = 0.0f;
    if (mx == -INFINITY) {
#pragma unroll
        for (int j = 0; j < KNN; ++j) w[j] = 0.0f;   // nan_to_num(softmax(all -inf)) == 0
    } else {
#pragma unroll
        for (int j = 0; j < KNN; ++j) { w[j] = __expf(logit[j] - mx); s += w[j]; }
        const float inv = 1.0f / s;
#pragma unroll
        for (int j = 0; j < KNN; ++j) w[j] *= inv;
    }
    float o0 = 0.0f, o1 = 0.0f;
#pragma unroll
    for (int j = 0; j < KNN; ++j) {
        const float* vp = qkv + (size_t)(b * N + nb[j]) * QKVW + 2 * DD + h * HEAD_DIM;
        o0 += w[j] * vp[2 * lane];
        o1 += w[j] * vp[2 * lane + 1];
    }
    o0 = 0.5f * o0 * (1.0f + erff(o0 * 0.70710678118654752f));
    o1 = 0.5f * o1 * (1.0f + erff(o1 * 0.70710678118654752f));
    __bf16* gp = gout + (size_t)node * DD + h * HEAD_DIM + 2 * lane;
    gp[0] = (__bf16)o0;
    gp[1] = (__bf16)o1;
}

// GEMM 2: out = x + G(16384x256)*Wproj^T + bproj
__global__ void __launch_bounds__(128) gemm_proj_kernel(const __bf16* __restrict__ G,
                                                        const __bf16* __restrict__ W,
                                                        const float* __restrict__ Xres,
                                                        const float* __restrict__ bias,
                                                        float* __restrict__ Y) {
    __shared__ __bf16 sW[64 * WSTRIDE];
    const int tid  = threadIdx.x;
    const int lane = tid & 31, wave = tid >> 5;
    const int half = lane >> 4, l16 = lane & 15;
    const int m0 = blockIdx.x * 128 + wave * 32;
    const int n0 = blockIdx.y * 64;

    stage_w_async(W, n0, sW, tid);

    v8f acc[2][4] = {};
    const __bf16* gr0 = G + (size_t)(m0 + l16) * DD;
    const __bf16* gr1 = G + (size_t)(m0 + 16 + l16) * DD;

    for (int k0 = 0; k0 < DD; k0 += 32) {
        if (k0 + 32 < DD) {
            __builtin_prefetch(gr0 + k0 + 32 + half * 8, 0, 0);
            __builtin_prefetch(gr1 + k0 + 32 + half * 8, 0, 0);
        }
        v16bf a0 = a_frag(gr0, k0, half);
        v16bf a1 = a_frag(gr1, k0, half);
        const int kb = k0 + half * 16;
#pragma unroll
        for (int t = 0; t < 4; ++t) {
            v16bf b = lds_b_frag(sW, t * 16 + l16, kb);
            acc[0][t] = __builtin_amdgcn_wmma_f32_16x16x32_bf16(false, a0, false, b, (short)0, acc[0][t], false, false);
            acc[1][t] = __builtin_amdgcn_wmma_f32_16x16x32_bf16(false, a1, false, b, (short)0, acc[1][t], false, false);
        }
    }
#pragma unroll
    for (int s = 0; s < 2; ++s)
#pragma unroll
        for (int v = 0; v < 8; ++v) {
            const size_t r = (size_t)(m0 + s * 16 + half * 8 + v) * DD + n0 + l16;
#pragma unroll
            for (int t = 0; t < 4; ++t)
                Y[r + t * 16] = Xres[r + t * 16] + acc[s][t][v] + bias[n0 + l16 + t * 16];
        }
}

// ---------------- host-side launcher ----------------
extern "C" void kernel_launch(void* const* d_in, const int* in_sizes, int n_in,
                              void* d_out, int out_size, void* d_ws, size_t ws_size,
                              hipStream_t stream) {
    const float*         x      = (const float*)d_in[0];         // (B,N,D) f32
    const float*         coords = (const float*)d_in[1];         // (B,N,2) f32
    const unsigned char* mask   = (const unsigned char*)d_in[2]; // (B,N) bool (1B)
    const float*         Wqkv   = (const float*)d_in[3];         // (768,256) f32
    const float*         Wproj  = (const float*)d_in[4];         // (256,256) f32
    const float*         bproj  = (const float*)d_in[5];         // (256,) f32
    float*               out    = (float*)d_out;

    // workspace layout (256B aligned)
    char* ws = (char*)d_ws;
    int*    nn    = (int*)(ws + 0);                       //  589,824 B
    __bf16* xbf   = (__bf16*)(ws + 589824);               // 8,388,608 B
    __bf16* wqb   = (__bf16*)(ws + 8978432);              //   393,216 B
    __bf16* wpb   = (__bf16*)(ws + 9371648);              //   131,072 B
    float*  qkv   = (float*)(ws + 9502720);               // 50,331,648 B
    __bf16* gbf   = (__bf16*)(ws + 59834368);             // 8,388,608 B

    // 1) bf16 conversions
    {
        int n8 = RTOT * DD / 8;                    // 524,288
        cvt_bf16_kernel<<<(n8 + 255) / 256, 256, 0, stream>>>(x, xbf, n8);
        n8 = QKVW * DD / 8;                        // 24,576
        cvt_bf16_kernel<<<(n8 + 255) / 256, 256, 0, stream>>>(Wqkv, wqb, n8);
        n8 = DD * DD / 8;                          // 8,192
        cvt_bf16_kernel<<<(n8 + 255) / 256, 256, 0, stream>>>(Wproj, wpb, n8);
    }
    // 2) kNN graph
    knn_kernel<<<dim3(NN / 256, BB), 256, 0, stream>>>(coords, mask, nn, NN);
    // 3) QKV projection (WMMA bf16, async-LDS staged weights)
    gemm_qkv_kernel<<<dim3(RTOT / 128, QKVW / 64), 128, 0, stream>>>(xbf, wqb, qkv);
    // 4) neighbor attention + GELU -> bf16
    attn_kernel<<<RTOT, 128, 0, stream>>>(qkv, nn, mask, gbf, NN);
    // 5) output projection + bias + residual (WMMA bf16, async-LDS staged weights)
    gemm_proj_kernel<<<dim3(RTOT / 128, DD / 64), 128, 0, stream>>>(gbf, wpb, x, bproj, out);
}
